// MarketRegimeAdapter_40638980554832
// MI455X (gfx1250) — matrix operand order repaired and checked by reading, not compiled, not run on hardware
//
#include <hip/hip_runtime.h>
#include <math.h>

typedef __attribute__((ext_vector_type(16))) __bf16        v16bf;
typedef __attribute__((ext_vector_type(8)))  float         v8f;
typedef __attribute__((ext_vector_type(8)))  unsigned int  v8u;

#define LN_EPS 1e-5f

// ---------- helpers ----------
__device__ __forceinline__ unsigned short bf16_rne(float f) {
    unsigned u = __float_as_uint(f);
    u += 0x7FFFu + ((u >> 16) & 1u);          // round-to-nearest-even to bf16
    return (unsigned short)(u >> 16);
}
__device__ __forceinline__ float bf16f(unsigned short h) {
    return __uint_as_float(((unsigned)h) << 16);
}
__device__ __forceinline__ float gelu_exact(float x) {
    return 0.5f * x * (1.0f + erff(x * 0.70710678118654752f));
}
__device__ __forceinline__ v8f wmma_bf16(v16bf a, v16bf b, v8f c) {
    // D = A(16x32 bf16) * B(32x16 bf16) + C(16x16 f32)
    return __builtin_amdgcn_wmma_f32_16x16x32_bf16(
        /*neg_a=*/false, a, /*neg_b=*/false, b,
        /*c_mod=*/(short)0, c, /*reuse_a=*/false, /*reuse_b=*/false);
}

// =====================================================================
// Kernel 0: pack W1/W2 (fp32 [8][512][512]) into WMMA B-fragment layout,
// split into bf16 hi/lo. Fragment = (r, ntile, kchunk): 256 dwords hi +
// 256 dwords lo; dword (lane,l, vgpr v) at offset lane*8+v so each lane's
// fragment is 32 contiguous bytes.
// B 32x16 bf16 order used here (A matches): lane<16: n=lane,  k=2v,2v+1
//                      lane>=16: n=lane-16, k=16+2v,16+2v+1
// =====================================================================
__global__ void pack_weights_kernel(const float* __restrict__ W1,
                                    const float* __restrict__ W2,
                                    unsigned* __restrict__ packed) {
    int bi    = blockIdx.x;          // 0..8191 (4096 fragments per matrix)
    int which = bi >> 12;
    int rem   = bi & 4095;
    int r     = rem >> 9;            // regime 0..7
    int nt    = (rem >> 4) & 31;     // n tile 0..31
    int kc    = rem & 15;            // k chunk 0..15
    const float* W = which ? W2 : W1;
    unsigned* dst = packed + (size_t)which * 2097152u
                  + ((size_t)((r * 32 + nt) * 16 + kc)) * 512u;
    int tid = threadIdx.x;           // 256 threads: one dword pair each
    int l = tid >> 3, v = tid & 7;
    int n = nt * 16 + (l & 15);
    int k = kc * 32 + ((l < 16) ? 0 : 16) + v * 2;
    const float* s = W + ((size_t)(r * 512 + k)) * 512 + n;
    float w0 = s[0], w1 = s[512];
    unsigned short h0 = bf16_rne(w0), h1 = bf16_rne(w1);
    unsigned short l0 = bf16_rne(w0 - bf16f(h0));
    unsigned short l1 = bf16_rne(w1 - bf16f(h1));
    dst[tid]        = (unsigned)h0 | ((unsigned)h1 << 16);
    dst[256 + tid]  = (unsigned)l0 | ((unsigned)l1 << 16);
}

// =====================================================================
// Kernel 1: summary[b][c] = mean over 4096 rows of features[b][row][c]
// =====================================================================
__global__ void summary_kernel(const float* __restrict__ feat,
                               float* __restrict__ summary) {
    int b = blockIdx.x;
    int c = blockIdx.y * 256 + threadIdx.x;
    const float* p = feat + (size_t)b * 4096 * 512 + c;
    float s = 0.f;
    #pragma unroll 8
    for (int row = 0; row < 4096; ++row) s += p[(size_t)row * 512];
    summary[b * 512 + c] = s * (1.0f / 4096.0f);
}

// =====================================================================
// Kernel 2: regime[b] = argmax( (gelu(summary@Wc1+bc1)) @ Wc2 + bc2 )
// (softmax is monotone -> argmax over logits)
// =====================================================================
__global__ void classify_kernel(const float* __restrict__ summary,
                                const float* __restrict__ Wc1,
                                const float* __restrict__ bc1,
                                const float* __restrict__ Wc2,
                                const float* __restrict__ bc2,
                                int* __restrict__ regime) {
    __shared__ float h[16 * 256];
    __shared__ float logits[16 * 8];
    int t = threadIdx.x;             // 256 threads
    for (int b = 0; b < 16; ++b) {
        float s = bc1[t];
        for (int k = 0; k < 512; ++k) s += summary[b * 512 + k] * Wc1[k * 256 + t];
        h[b * 256 + t] = gelu_exact(s);
    }
    __syncthreads();
    if (t < 128) {
        int b = t >> 3, rr = t & 7;
        float s = bc2[rr];
        for (int k = 0; k < 256; ++k) s += h[b * 256 + k] * Wc2[k * 8 + rr];
        logits[b * 8 + rr] = s;
    }
    __syncthreads();
    if (t < 16) {
        float best = logits[t * 8];
        int bi = 0;
        for (int rr = 1; rr < 8; ++rr) {
            float v = logits[t * 8 + rr];
            if (v > best) { best = v; bi = rr; }
        }
        regime[t] = bi;
    }
}

// =====================================================================
// Kernel 3: fused  X@W1 -> +b1 -> GELU -> @W2 -> +b2 -> LayerNorm -> *g+b
// One workgroup = 64 rows x all 512 cols of one batch. 8 waves in 4Mx2N.
// bf16x3 split-precision WMMA, fp32 accumulate. Intermediate y in LDS
// (pitch 516 floats to avoid 64-bank conflicts).
// =====================================================================
#define YPITCH 516

__global__ void __launch_bounds__(256)
adapter_kernel(const float* __restrict__ feat,
               const unsigned* __restrict__ packedW,
               const float* __restrict__ b1, const float* __restrict__ b2,
               const float* __restrict__ gamma, const float* __restrict__ beta,
               const int* __restrict__ regime,
               float* __restrict__ out) {
    extern __shared__ float smem[];
    float* ybuf = smem;                  // 64 * 516
    float* red  = smem + 64 * YPITCH;    // 256
    float* red2 = red + 256;             // 256
    float* mu_s = red2 + 256;            // 64
    float* rs_s = mu_s + 64;             // 64

    const int mt   = blockIdx.x;         // M tile 0..63
    const int b    = blockIdx.y;         // batch 0..15
    const int tid  = threadIdx.x;
    const int lane = tid & 31;
    const int wave = tid >> 5;
    const int wm   = wave >> 1;          // 0..3 (M strip)
    const int wn   = wave & 1;           // 0..1 (N half)
    const int rbase = wm * 16;
    const int cbase = wn * 256;
    const int hl    = (lane < 16) ? 0 : 8;   // C row-half; A k-offset is 2*hl
    const int ln16  = lane & 15;

    const int r = regime[b];
    const unsigned* w1pk = packedW + (size_t)r * 262144u;
    const unsigned* w2pk = packedW + 2097152u + (size_t)r * 262144u;
    const float* b1r = b1    + r * 512;
    const float* b2r = b2    + r * 512;
    const float* gr  = gamma + r * 512;
    const float* br  = beta  + r * 512;

    v8f acc[16];

    // ---------------- GEMM1: y = gelu(X @ W1 + b1) ----------------
    #pragma unroll
    for (int t = 0; t < 16; ++t) acc[t] = (v8f)0.0f;

    const float* Xrow =
        feat + ((size_t)b * 4096 + (size_t)mt * 64 + rbase + ln16) * 512;

    for (int kc = 0; kc < 16; ++kc) {
        const int kb = kc * 32 + 2 * hl;
        float4 f0 = *(const float4*)(Xrow + kb);
        float4 f1 = *(const float4*)(Xrow + kb + 4);
        float4 f2 = *(const float4*)(Xrow + kb + 8);
        float4 f3 = *(const float4*)(Xrow + kb + 12);
        float xs[16] = {f0.x, f0.y, f0.z, f0.w, f1.x, f1.y, f1.z, f1.w,
                        f2.x, f2.y, f2.z, f2.w, f3.x, f3.y, f3.z, f3.w};
        v8u uhi, ulo;
        #pragma unroll
        for (int i = 0; i < 8; ++i) {
            float a = xs[2 * i], c2 = xs[2 * i + 1];
            unsigned short ha = bf16_rne(a), hb = bf16_rne(c2);
            uhi[i] = (unsigned)ha | ((unsigned)hb << 16);
            unsigned short la = bf16_rne(a  - bf16f(ha));
            unsigned short lb = bf16_rne(c2 - bf16f(hb));
            ulo[i] = (unsigned)la | ((unsigned)lb << 16);
        }
        v16bf Ahi = __builtin_bit_cast(v16bf, uhi);
        v16bf Alo = __builtin_bit_cast(v16bf, ulo);
        const unsigned* fbase = w1pk + (size_t)kc * 512 + lane * 8;
        #pragma unroll
        for (int t = 0; t < 16; ++t) {
            int nt = wn * 16 + t;
            const unsigned* fp = fbase + (size_t)nt * (16 * 512);
            v16bf Bhi = __builtin_bit_cast(v16bf, *(const v8u*)fp);
            v16bf Blo = __builtin_bit_cast(v16bf, *(const v8u*)(fp + 256));
            acc[t] = wmma_bf16(Ahi, Bhi, acc[t]);
            acc[t] = wmma_bf16(Alo, Bhi, acc[t]);
            acc[t] = wmma_bf16(Ahi, Blo, acc[t]);
        }
    }

    // epilogue 1: bias + exact GELU -> LDS
    #pragma unroll
    for (int t = 0; t < 16; ++t) {
        int n = cbase + t * 16 + ln16;
        float bias = b1r[n];
        #pragma unroll
        for (int v = 0; v < 8; ++v) {
            int m = rbase + hl + v;
            ybuf[m * YPITCH + n] = gelu_exact(acc[t][v] + bias);
        }
    }
    __syncthreads();

    // ---------------- GEMM2: z = y @ W2 + b2 ----------------
    #pragma unroll
    for (int t = 0; t < 16; ++t) acc[t] = (v8f)0.0f;

    const float* Yrow = ybuf + (rbase + ln16) * YPITCH;

    for (int kc = 0; kc < 16; ++kc) {
        const int kb = kc * 32 + 2 * hl;
        float4 f0 = *(const float4*)(Yrow + kb);
        float4 f1 = *(const float4*)(Yrow + kb + 4);
        float4 f2 = *(const float4*)(Yrow + kb + 8);
        float4 f3 = *(const float4*)(Yrow + kb + 12);
        float xs[16] = {f0.x, f0.y, f0.z, f0.w, f1.x, f1.y, f1.z, f1.w,
                        f2.x, f2.y, f2.z, f2.w, f3.x, f3.y, f3.z, f3.w};
        v8u uhi, ulo;
        #pragma unroll
        for (int i = 0; i < 8; ++i) {
            float a = xs[2 * i], c2 = xs[2 * i + 1];
            unsigned short ha = bf16_rne(a), hb = bf16_rne(c2);
            uhi[i] = (unsigned)ha | ((unsigned)hb << 16);
            unsigned short la = bf16_rne(a  - bf16f(ha));
            unsigned short lb = bf16_rne(c2 - bf16f(hb));
            ulo[i] = (unsigned)la | ((unsigned)lb << 16);
        }
        v16bf Ahi = __builtin_bit_cast(v16bf, uhi);
        v16bf Alo = __builtin_bit_cast(v16bf, ulo);
        const unsigned* fbase = w2pk + (size_t)kc * 512 + lane * 8;
        #pragma unroll
        for (int t = 0; t < 16; ++t) {
            int nt = wn * 16 + t;
            const unsigned* fp = fbase + (size_t)nt * (16 * 512);
            v16bf Bhi = __builtin_bit_cast(v16bf, *(const v8u*)fp);
            v16bf Blo = __builtin_bit_cast(v16bf, *(const v8u*)(fp + 256));
            acc[t] = wmma_bf16(Ahi, Bhi, acc[t]);
            acc[t] = wmma_bf16(Alo, Bhi, acc[t]);
            acc[t] = wmma_bf16(Ahi, Blo, acc[t]);
        }
    }
    __syncthreads();   // everyone done reading y before we overwrite it

    // epilogue 2: +b2 -> LDS (pre-LayerNorm z)
    #pragma unroll
    for (int t = 0; t < 16; ++t) {
        int n = cbase + t * 16 + ln16;
        float bias = b2r[n];
        #pragma unroll
        for (int v = 0; v < 8; ++v) {
            int m = rbase + hl + v;
            ybuf[m * YPITCH + n] = acc[t][v] + bias;
        }
    }
    __syncthreads();

    // ---------------- LayerNorm (biased var) + gamma/beta ----------------
    {
        int rrow = tid >> 2, q = tid & 3;
        const float* rp = ybuf + rrow * YPITCH + q * 128;
        float s = 0.f, s2 = 0.f;
        #pragma unroll 4
        for (int j = 0; j < 128; ++j) { float v = rp[j]; s += v; s2 += v * v; }
        red[tid] = s; red2[tid] = s2;
    }
    __syncthreads();
    if (tid < 64) {
        float s  = red[tid*4]  + red[tid*4+1]  + red[tid*4+2]  + red[tid*4+3];
        float s2 = red2[tid*4] + red2[tid*4+1] + red2[tid*4+2] + red2[tid*4+3];
        float mu  = s * (1.0f / 512.0f);
        float var = s2 * (1.0f / 512.0f) - mu * mu;
        mu_s[tid] = mu;
        rs_s[tid] = rsqrtf(var + LN_EPS);
    }
    __syncthreads();
    float* outp = out + ((size_t)b * 4096 + (size_t)mt * 64) * 512;
    for (int idx = tid; idx < 64 * 512; idx += 256) {
        int rrow = idx >> 9, c = idx & 511;
        float v = ybuf[rrow * YPITCH + c];
        outp[(size_t)rrow * 512 + c] = (v - mu_s[rrow]) * rs_s[rrow] * gr[c] + br[c];
    }
}

// =====================================================================
extern "C" void kernel_launch(void* const* d_in, const int* in_sizes, int n_in,
                              void* d_out, int out_size, void* d_ws, size_t ws_size,
                              hipStream_t stream) {
    const float* feat  = (const float*)d_in[0];
    const float* W1    = (const float*)d_in[1];
    const float* b1    = (const float*)d_in[2];
    const float* W2    = (const float*)d_in[3];
    const float* b2    = (const float*)d_in[4];
    const float* gamma = (const float*)d_in[5];
    const float* beta  = (const float*)d_in[6];
    const float* Wc1   = (const float*)d_in[7];
    const float* bc1   = (const float*)d_in[8];
    const float* Wc2   = (const float*)d_in[9];
    const float* bc2   = (const float*)d_in[10];
    float* out = (float*)d_out;

    // workspace layout: [0,16MiB) packed bf16 hi/lo weights; then summary; regime
    unsigned* packed  = (unsigned*)d_ws;
    float*    summary = (float*)((char*)d_ws + (size_t)16 * 1024 * 1024);
    int*      regime  = (int*)((char*)d_ws + (size_t)16 * 1024 * 1024 + 32 * 1024);

    pack_weights_kernel<<<dim3(8192), dim3(256), 0, stream>>>(W1, W2, packed);
    summary_kernel<<<dim3(16, 2), dim3(256), 0, stream>>>(feat, summary);
    classify_kernel<<<dim3(1), dim3(256), 0, stream>>>(summary, Wc1, bc1, Wc2, bc2, regime);

    size_t smem = (size_t)(64 * YPITCH + 256 + 256 + 64 + 64) * sizeof(float);
    adapter_kernel<<<dim3(64, 16), dim3(256), smem, stream>>>(
        feat, packed, b1, b2, gamma, beta, regime, out);
}
